// Self_Attention_Layer_4724464025889
// MI455X (gfx1250) — matrix operand order repaired and checked by reading, not compiled, not run on hardware
//
#include <hip/hip_runtime.h>

// ---------------------------------------------------------------------------
// Self-attention for MI455X (gfx1250, wave32, WMMA bf16 -> f32 accumulate)
// B=16, S=2048, D_IN=256, D_ATT=128, scale = 1/sqrt(256) = 1/16
// Round 2: K/V tiles staged to LDS via global_load_async_to_lds_b128 (ASYNCcnt)
// ---------------------------------------------------------------------------

typedef __attribute__((ext_vector_type(16))) __bf16        v16bf;
typedef __attribute__((ext_vector_type(8)))  float         v8f;
typedef __attribute__((ext_vector_type(4)))  float         f32x4;
typedef __attribute__((ext_vector_type(4)))  unsigned int  u32x4;

union Frag16 { v16bf bf; u32x4 u[2]; };   // 32 bytes
union Pack8  { __bf16 h[8]; u32x4 u; };   // 16 bytes

#define WMMA_BF16(A, B, C) \
  __builtin_amdgcn_wmma_f32_16x16x32_bf16(false, (A), false, (B), (short)0, (C), false, false)

static __device__ __forceinline__ int lane_id() { return (int)(threadIdx.x & 31); }

// Async copy 16B global -> LDS, tracked by ASYNCcnt (CDNA5 ISA ch.10.7/15.18.3)
static __device__ __forceinline__ void async_copy_b128(unsigned int lds_addr,
                                                       const __bf16* gptr) {
  asm volatile("global_load_async_to_lds_b128 %0, %1, off"
               :: "v"(lds_addr), "v"((unsigned long long)(uintptr_t)gptr)
               : "memory");
}
static __device__ __forceinline__ void wait_async0() {
  asm volatile("s_wait_asynccnt 0" ::: "memory");
}

// B operand (32x16): element (k,n) = src[(n0+n)*stride + k0 + k]
// lane<16: N=lane, K={k0..k0+7, k0+16..k0+23}; lane>=16: N=lane-16, K={k0+8..15, k0+24..31}
static __device__ __forceinline__ v16bf load_bfrag(const __bf16* src, int stride, int n0, int k0) {
  const int lane = lane_id();
  const __bf16* p = src + (size_t)(n0 + (lane & 15)) * stride + k0 + ((lane & 16) ? 8 : 0);
  Frag16 f;
  f.u[0] = *(const u32x4*)(p);
  f.u[1] = *(const u32x4*)(p + 16);
  return f.bf;
}

// A operand (16x32) from row-major bf16: element (m,k) = src[(m0+m)*stride + k0 + k]
// lane<16: M=lane, K={k0..k0+7, k0+16..k0+23}; lane>=16: M=lane-16, K={k0+8..15, k0+24..31}
static __device__ __forceinline__ v16bf load_afrag_bf16(const __bf16* src, int stride, int m0, int k0) {
  const int lane = lane_id();
  const __bf16* p = src + (size_t)(m0 + (lane & 15)) * stride + k0 + ((lane & 16) ? 8 : 0);
  Frag16 f;
  f.u[0] = *(const u32x4*)(p);
  f.u[1] = *(const u32x4*)(p + 16);
  return f.bf;
}

// A operand (16x32) from row-major f32 with bf16 convert
static __device__ __forceinline__ v16bf load_afrag_f32(const float* src, int stride, int m0, int k0) {
  const int lane = lane_id();
  const float* p = src + (size_t)(m0 + (lane & 15)) * stride + k0 + ((lane & 16) ? 8 : 0);
  f32x4 a0 = *(const f32x4*)(p);
  f32x4 a1 = *(const f32x4*)(p + 4);
  f32x4 b0 = *(const f32x4*)(p + 16);
  f32x4 b1 = *(const f32x4*)(p + 20);
  v16bf r;
#pragma unroll
  for (int i = 0; i < 4; ++i) {
    r[i]      = (__bf16)a0[i];
    r[4 + i]  = (__bf16)a1[i];
    r[8 + i]  = (__bf16)b0[i];
    r[12 + i] = (__bf16)b1[i];
  }
  return r;
}

// Cross-lane reduction within each 16-lane half (rows of one C half live across 16 lanes)
template <int XM>
static __device__ __forceinline__ float swz_xor(float v) {
  return __int_as_float(__builtin_amdgcn_ds_swizzle(__float_as_int(v), 0x1f | (XM << 10)));
}
static __device__ __forceinline__ float half16_max(float v) {
  v = fmaxf(v, swz_xor<1>(v)); v = fmaxf(v, swz_xor<2>(v));
  v = fmaxf(v, swz_xor<4>(v)); v = fmaxf(v, swz_xor<8>(v));
  return v;
}
static __device__ __forceinline__ float half16_sum(float v) {
  v += swz_xor<1>(v); v += swz_xor<2>(v);
  v += swz_xor<4>(v); v += swz_xor<8>(v);
  return v;
}

// ---------------------------------------------------------------------------
// Kernel 0: transpose W (256x128 f32, k-major) -> WT[mat][n=0..127][k=0..255] bf16
// ---------------------------------------------------------------------------
__global__ void wt_kernel(const float* __restrict__ Wq, const float* __restrict__ Wk,
                          const float* __restrict__ Wv, __bf16* __restrict__ WT) {
  const int i = (int)(blockIdx.x * 256 + threadIdx.x);
  if (i >= 3 * 128 * 256) return;
  const int mat = i / (128 * 256);
  const int rem = i - mat * (128 * 256);
  const int n = rem >> 8;
  const int k = rem & 255;
  const float* W = (mat == 0) ? Wq : ((mat == 1) ? Wk : Wv);
  WT[i] = (__bf16)W[k * 128 + n];
}

// ---------------------------------------------------------------------------
// Kernel 1: QKV projection. One wave = 16 rows x 128 cols, K=256 (8 WMMA steps x 8 n-tiles).
// Q,K written row-major bf16. V written transposed: VT[b][d][s] bf16 (packed 16B stores).
// grid = (512, 3), block = 128 (4 waves)
// ---------------------------------------------------------------------------
__global__ __launch_bounds__(128) void qkv_kernel(const float* __restrict__ X,
                                                  const __bf16* __restrict__ WT,
                                                  __bf16* __restrict__ Qb,
                                                  __bf16* __restrict__ Kb,
                                                  __bf16* __restrict__ VTb) {
  const int wave = (int)(threadIdx.x >> 5);
  const int lane = lane_id();
  const int nlo  = lane & 15;
  const int hi   = lane >> 4;
  const int m0   = ((int)blockIdx.x * 4 + wave) * 16;   // global row tile (B*S rows)
  const int mat  = (int)blockIdx.y;                     // 0=Q 1=K 2=V
  const __bf16* wt = WT + mat * (128 * 256);

  v8f acc[8] = {};
#pragma unroll
  for (int ks = 0; ks < 8; ++ks) {
    const v16bf a = load_afrag_f32(X, 256, m0, ks * 32);
#pragma unroll
    for (int nt = 0; nt < 8; ++nt) {
      const v16bf b = load_bfrag(wt, 256, nt * 16, ks * 32);
      acc[nt] = WMMA_BF16(a, b, acc[nt]);
    }
  }

  // C layout: VGPR r -> (M = r + 8*hi, N = nlo) within the 16x16 tile
  if (mat < 2) {
    __bf16* outp = (mat == 0) ? Qb : Kb;
#pragma unroll
    for (int nt = 0; nt < 8; ++nt)
#pragma unroll
      for (int r = 0; r < 8; ++r)
        outp[(size_t)(m0 + r + 8 * hi) * 128 + nt * 16 + nlo] = (__bf16)acc[nt][r];
  } else {
    const int bIdx  = m0 >> 11;                 // S = 2048
    const int sBase = (m0 & 2047) + 8 * hi;     // 8 consecutive s per lane -> one 16B store
#pragma unroll
    for (int nt = 0; nt < 8; ++nt) {
      Pack8 p;
#pragma unroll
      for (int r = 0; r < 8; ++r) p.h[r] = (__bf16)acc[nt][r];
      *(u32x4*)(VTb + (size_t)(bIdx * 128 + nt * 16 + nlo) * 2048 + sBase) = p.u;
    }
  }
}

// ---------------------------------------------------------------------------
// Kernel 2: flash attention. One wave = 16 queries, full D_ATT=128.
// 64-key blocks; per block the K tile (64x128 bf16) and V tile (128x64 bf16, from
// VT) are staged to LDS once per workgroup with async copies, shared by 4 waves.
// LDS strides padded (136 / 72 elems) for conflict-free 16-lane fragment reads.
// grid = 512, block = 128 (4 waves); all waves in a block share a batch -> uniform flow.
// ---------------------------------------------------------------------------
__global__ __launch_bounds__(128) void attn_kernel(const __bf16* __restrict__ Qb,
                                                   const __bf16* __restrict__ Kb,
                                                   const __bf16* __restrict__ VT,
                                                   const int* __restrict__ sen_len,
                                                   float* __restrict__ out) {
  __shared__ __bf16 Klds[64 * 136];    // 17408 B: 64 keys x 128 d (stride 136)
  __shared__ __bf16 Vlds[128 * 72];    // 18432 B: 128 d x 64 keys (stride 72)
  __shared__ __bf16 plds[4][16 * 64];  //  8192 B: per-wave P staging

  const int tid  = (int)threadIdx.x;
  const int wave = tid >> 5;
  const int lane = lane_id();
  const int nlo  = lane & 15;
  const int hi   = lane >> 4;

  const int tile = (int)blockIdx.x * 4 + wave;  // 16-query tile id, 0..2047
  const int bIdx = tile >> 7;                   // 128 tiles per batch
  const int q0   = (tile & 127) << 4;           // query base within batch

  const int len = sen_len[bIdx];
  const __bf16* Qrow = Qb + (size_t)bIdx * 2048 * 128;
  const __bf16* Krow = Kb + (size_t)bIdx * 2048 * 128;
  const __bf16* Vt   = VT + (size_t)bIdx * 128 * 2048;
  __bf16* myp = plds[wave];

  // LDS byte offsets (generic shared pointer truncates to LDS offset per aperture rule)
  const unsigned int kbase = (unsigned int)(uintptr_t)&Klds[0];
  const unsigned int vbase = (unsigned int)(uintptr_t)&Vlds[0];

  // Preload Q fragments (K = 128 -> 4 steps of 32), kept in registers for the whole sweep
  v16bf qa[4];
#pragma unroll
  for (int ks = 0; ks < 4; ++ks) qa[ks] = load_afrag_bf16(Qrow, 128, q0, ks * 32);

  v8f z[8] = {};
  float rowM[8], rowS[8];
#pragma unroll
  for (int r = 0; r < 8; ++r) { rowM[r] = -__builtin_inff(); rowS[r] = 0.0f; }

  const int nblocks = (len + 63) >> 6;
  for (int kb = 0; kb < nblocks; ++kb) {
    const int key0 = kb << 6;

    // ---- stage K tile: 64 rows x 256B = 1024 x 16B segments, coalesced ----
#pragma unroll
    for (int i = 0; i < 8; ++i) {
      const int g   = tid + 128 * i;   // 0..1023
      const int row = g >> 4;          // key within block
      const int c   = g & 15;          // 16B segment within row
      async_copy_b128(kbase + (unsigned int)(row * 272 + c * 16),
                      Krow + (size_t)(key0 + row) * 128 + c * 8);
    }
    // ---- stage V tile: 128 rows (d) x 128B = 1024 x 16B segments ----
#pragma unroll
    for (int i = 0; i < 8; ++i) {
      const int g   = tid + 128 * i;
      const int row = g >> 3;          // d
      const int c   = g & 7;
      async_copy_b128(vbase + (unsigned int)(row * 144 + c * 16),
                      Vt + (size_t)row * 2048 + key0 + c * 8);
    }
    wait_async0();
    __syncthreads();

    // S_tile(16q x 64k) = Q(16x128) @ K^T, B-fragments from LDS (K rows are K-contiguous)
    v8f s[4] = {};
#pragma unroll
    for (int ks = 0; ks < 4; ++ks) {
#pragma unroll
      for (int nt = 0; nt < 4; ++nt) {
        const v16bf bk = load_bfrag(Klds, 136, nt * 16, ks * 32);
        s[nt] = WMMA_BF16(qa[ks], bk, s[nt]);
      }
    }

    // mask + scale: valid -> s/16, masked -> PADDING_VAL/16 = -2^28 (per-lane cndmask)
#pragma unroll
    for (int nt = 0; nt < 4; ++nt) {
      const bool valid = (key0 + nt * 16 + nlo) < len;
#pragma unroll
      for (int r = 0; r < 8; ++r)
        s[nt][r] = valid ? s[nt][r] * 0.0625f : -2.68435456e8f;
    }

    // online softmax: rows of each C half are striped across 16 lanes
#pragma unroll
    for (int r = 0; r < 8; ++r) {
      float bm = fmaxf(fmaxf(s[0][r], s[1][r]), fmaxf(s[2][r], s[3][r]));
      bm = half16_max(bm);
      const float nm = fmaxf(rowM[r], bm);
      const float sc = __expf(rowM[r] - nm);   // first block: exp(-inf) = 0
      rowM[r] = nm;
      rowS[r] *= sc;
#pragma unroll
      for (int dt = 0; dt < 8; ++dt) z[dt][r] *= sc;
      float ps = 0.0f;
#pragma unroll
      for (int nt = 0; nt < 4; ++nt) {
        const float e = __expf(s[nt][r] - nm);
        s[nt][r] = e;
        ps += e;
      }
      rowS[r] += half16_sum(ps);
    }

    // P (C-layout) -> LDS row-major bf16 [16][64]
#pragma unroll
    for (int nt = 0; nt < 4; ++nt)
#pragma unroll
      for (int r = 0; r < 8; ++r)
        myp[(r + 8 * hi) * 64 + nt * 16 + nlo] = (__bf16)s[nt][r];

    __syncthreads();

    // Z += P(16x64) @ V(64x128): V^T rows in LDS are key-contiguous -> direct B-fragments
#pragma unroll
    for (int ks2 = 0; ks2 < 2; ++ks2) {
      const v16bf pa = load_afrag_bf16(myp, 64, 0, ks2 * 32);
#pragma unroll
      for (int dt = 0; dt < 8; ++dt) {
        const v16bf bv = load_bfrag(Vlds, 72, dt * 16, ks2 * 32);
        z[dt] = WMMA_BF16(pa, bv, z[dt]);
      }
    }
    __syncthreads();   // protect staged tiles until all waves finish reading
  }

  // finalize: divide by row sums, store f32 (coalesced across the 16 lanes of each half)
#pragma unroll
  for (int r = 0; r < 8; ++r) {
    const float inv = 1.0f / rowS[r];
    const size_t rowOff = (size_t)(bIdx * 2048 + q0 + r + 8 * hi) * 128;
#pragma unroll
    for (int dt = 0; dt < 8; ++dt)
      out[rowOff + dt * 16 + nlo] = z[dt][r] * inv;
  }
}

// ---------------------------------------------------------------------------
// Launch: ws layout = Qb(8MB) | Kb(8MB) | VT(8MB) | WT(192KB)
// ---------------------------------------------------------------------------
extern "C" void kernel_launch(void* const* d_in, const int* in_sizes, int n_in,
                              void* d_out, int out_size, void* d_ws, size_t ws_size,
                              hipStream_t stream) {
  const float* X   = (const float*)d_in[0];
  const int*   sen = (const int*)d_in[1];
  const float* Wq  = (const float*)d_in[2];
  const float* Wk  = (const float*)d_in[3];
  const float* Wv  = (const float*)d_in[4];
  float* out = (float*)d_out;

  char* ws = (char*)d_ws;
  __bf16* Qb  = (__bf16*)(ws);
  __bf16* Kb  = (__bf16*)(ws + (size_t)8 * 1024 * 1024);
  __bf16* VTb = (__bf16*)(ws + (size_t)16 * 1024 * 1024);
  __bf16* WT  = (__bf16*)(ws + (size_t)24 * 1024 * 1024);

  wt_kernel<<<(3 * 128 * 256 + 255) / 256, 256, 0, stream>>>(Wq, Wk, Wv, WT);

  dim3 g1(512, 3);           // 2048 row tiles / 4 waves per block, x3 matrices
  qkv_kernel<<<g1, 128, 0, stream>>>(X, WT, Qb, Kb, VTb);

  attn_kernel<<<512, 128, 0, stream>>>(Qb, Kb, VTb, sen, out);
}
